// VGG_tiny_MixQ_20624432956078
// MI455X (gfx1250) — matrix operand order repaired and checked by reading, not compiled, not run on hardware
//
#include <hip/hip_runtime.h>
#include <hip/hip_bf16.h>

typedef __attribute__((ext_vector_type(16))) _Float16 v16h;
typedef __attribute__((ext_vector_type(8)))  float    v8f;

#define NB 7  // BITS = 2..8 -> 7 mixture components

// ---------------------------------------------------------------- helpers ---

__device__ inline void softmax7(const float* __restrict__ a, float* s) {
  float mx = a[0];
#pragma unroll
  for (int i = 1; i < NB; ++i) mx = fmaxf(mx, a[i]);
  float sum = 0.f;
#pragma unroll
  for (int i = 0; i < NB; ++i) { float e = expf(a[i] - mx); s[i] = e; sum += e; }
  float inv = 1.f / sum;
#pragma unroll
  for (int i = 0; i < NB; ++i) s[i] *= inv;
}

__device__ inline float mix_quant_act(float x, const float* s) {
  float xc = fminf(fmaxf(x, 0.f), 1.f);
  float o = 0.f;
#pragma unroll
  for (int i = 0; i < NB; ++i) {
    float n = (float)((4 << i) - 1);          // 2^(i+2)-1 : 3,7,...,255
    o += s[i] * (rintf(xc * n) / n);          // rintf == round-half-even
  }
  return o;
}

// ------------------------------------------------------- weight quantizers ---

__global__ void __launch_bounds__(256) wtanh_max_k(const float* __restrict__ w,
                                                   int n, float* __restrict__ out) {
  float m = 0.f;
  for (int i = threadIdx.x; i < n; i += 256) m = fmaxf(m, fabsf(tanhf(w[i])));
  __shared__ float sh[256];
  sh[threadIdx.x] = m; __syncthreads();
  for (int off = 128; off > 0; off >>= 1) {
    if (threadIdx.x < off) sh[threadIdx.x] = fmaxf(sh[threadIdx.x], sh[threadIdx.x + off]);
    __syncthreads();
  }
  if (threadIdx.x == 0) out[0] = sh[0];
}

// DoReFa mixed-bitwidth weight quant -> f16 A matrix [Co][Kpad].
// K-order: k = r*Cip + ci  (tap-major, channel-minor), zero padded.
__global__ void __launch_bounds__(256) wquant_mix_k(const float* __restrict__ w,
                                                    const float* __restrict__ alpha,
                                                    const float* __restrict__ mxp,
                                                    _Float16* __restrict__ Aq,
                                                    int Ci, int Cip, int ciShift,
                                                    int Kpad, int Co) {
  __shared__ float s[NB + 1];
  if (threadIdx.x == 0) softmax7(alpha, s);
  __syncthreads();
  float inv2m = 0.5f / mxp[0];
  int total = Co * Kpad;
  for (int idx = blockIdx.x * 256 + threadIdx.x; idx < total; idx += gridDim.x * 256) {
    int co = idx / Kpad, kk = idx - co * Kpad;
    int r = kk >> ciShift, ci = kk & (Cip - 1);
    float o = 0.f;
    if (r < 9 && ci < Ci) {
      float t = tanhf(w[((size_t)co * Ci + ci) * 9 + r]) * inv2m + 0.5f;
#pragma unroll
      for (int i = 0; i < NB; ++i) {
        float n = (float)((4 << i) - 1);
        o += s[i] * (2.f * (rintf(t * n) / n) - 1.f);
      }
    }
    Aq[idx] = (_Float16)o;
  }
}

__global__ void __launch_bounds__(256) wquant8_k(const float* __restrict__ w,
                                                 const float* __restrict__ mxp,
                                                 _Float16* __restrict__ out, int n) {
  float inv2m = 0.5f / mxp[0];
  for (int idx = blockIdx.x * 256 + threadIdx.x; idx < n; idx += gridDim.x * 256) {
    float t = tanhf(w[idx]) * inv2m + 0.5f;
    out[idx] = (_Float16)(2.f * (rintf(t * 255.f) * (1.f / 255.f)) - 1.f);
  }
}

// -------------------------------------------------------- activation quant ---

// layer-0 input: NCHW f32 -> padded NHWC f16 [B][34][34][4]; halo & ch3 = 0
__global__ void __launch_bounds__(256) input_quant_k(const float* __restrict__ x,
                                                     const float* __restrict__ alpha,
                                                     _Float16* __restrict__ out,
                                                     int Bn) {
  __shared__ float s[NB + 1];
  if (threadIdx.x == 0) softmax7(alpha, s);
  __syncthreads();
  int total = Bn * 34 * 34 * 4;
  for (int idx = blockIdx.x * 256 + threadIdx.x; idx < total; idx += gridDim.x * 256) {
    int c = idx & 3;  int t = idx >> 2;
    int xp = t % 34;  int t2 = t / 34;
    int yp = t2 % 34; int b = t2 / 34;
    float o = 0.f;
    if (c < 3 && yp >= 1 && yp <= 32 && xp >= 1 && xp <= 32) {
      float xv = x[(((size_t)b * 3 + c) * 32 + (yp - 1)) * 32 + (xp - 1)];
      o = mix_quant_act(xv, s);
    }
    out[idx] = (_Float16)o;
  }
}

// ------------------------------------------------------------ WMMA conv ------

// One wave computes 16(M=Cout) x 64(N=B*H*W) via V_WMMA_F32_16X16X32_F16.
// act: padded NHWC [B][H+2][W+2][Cip]; wq: [Co][Kpad] with k = r*Cip + ci.
template <bool VEC>
__global__ void __launch_bounds__(32) conv3x3_wmma_k(const _Float16* __restrict__ act,
                                                     const _Float16* __restrict__ wq,
                                                     float* __restrict__ out,
                                                     int logW, int logH,
                                                     int Cip, int ciShift,
                                                     int K, int Kpad, int Nn) {
  const int lane = threadIdx.x & 31;
  const int col  = lane & 15;
  const int hi   = lane >> 4;
  const int W = 1 << logW, H = 1 << logH;
  const int Wp = W + 2, Hp = H + 2;
  const int n0 = blockIdx.x * 64;
  const _Float16* Arow = wq + (size_t)(blockIdx.y * 16 + col) * Kpad;

  int pbase[4];                       // spatial offset of padded center, per sub-tile
#pragma unroll
  for (int s = 0; s < 4; ++s) {
    int nc = n0 + 16 * s + col;
    int xx = nc & (W - 1);
    int yy = (nc >> logW) & (H - 1);
    int bb = nc >> (logW + logH);
    pbase[s] = (bb * Hp + yy + 1) * Wp + (xx + 1);
  }

  v8f c[4] = {};
  for (int kb = 0; kb < K; kb += 32) {
    // A fragment: halves 0..15 <- Arow[kb+16hi .. kb+16hi+15]
    v16h a;
    {
      union { uint4 u[2]; v16h v; } t;
      t.u[0] = *reinterpret_cast<const uint4*>(Arow + kb + 16 * hi);
      t.u[1] = *reinterpret_cast<const uint4*>(Arow + kb + 16 * hi + 8);
      a = t.v;
    }
    if (kb + 32 < Kpad) __builtin_prefetch(Arow + kb + 32, 0, 1);

    if (VEC) {
      // tap index r uniform over this K-block (kb multiple of 32, Cip >= 64)
      int r   = kb >> ciShift;
      int cib = kb & (Cip - 1);
      int q3  = (r * 86) >> 8;                    // r/3 for r in [0,8]
      int roff = (q3 - 1) * Wp + (r - 3 * q3 - 1);
#pragma unroll
      for (int s = 0; s < 4; ++s) {
        const _Float16* bp = act + (size_t)(pbase[s] + roff) * Cip + cib + 16 * hi;
        // 32B-aligned: (pbase+roff)*Cip is >=128B aligned, cib a multiple of 32, 16*hi
        v16h b = *reinterpret_cast<const v16h*>(bp);
        c[s] = __builtin_amdgcn_wmma_f32_16x16x32_f16(false, a, false, b,
                                                      (short)0, c[s], false, false);
      }
    } else {
      // layer-0 path (Cip=4): 4 aligned b64 group loads per fragment.
      // group g covers k = kb+16hi+4g .. +3 -> one tap r, full ci range 0..3.
      int rbase = (kb + 16 * hi) >> 2;
#pragma unroll
      for (int s = 0; s < 4; ++s) {
        union { uint2 u[4]; v16h v; } t;
#pragma unroll
        for (int g = 0; g < 4; ++g) {
          int r = rbase + g; r = r > 8 ? 8 : r;   // clamped rows have A == 0
          int q3 = (r * 86) >> 8;
          int roff = (q3 - 1) * Wp + (r - 3 * q3 - 1);
          t.u[g] = *reinterpret_cast<const uint2*>(act + (size_t)(pbase[s] + roff) * 4);
        }
        c[s] = __builtin_amdgcn_wmma_f32_16x16x32_f16(false, a, false, t.v,
                                                      (short)0, c[s], false, false);
      }
    }
  }

  // store D: VGPR r holds M = r + 8*hi, N = col ; out is [Co][B*H*W] f32
#pragma unroll
  for (int s = 0; s < 4; ++s) {
#pragma unroll
    for (int r = 0; r < 8; ++r) {
      int m = blockIdx.y * 16 + r + 8 * hi;
      out[(size_t)m * Nn + n0 + 16 * s + col] = c[s][r];
    }
  }
}

// --------------------------------------------------------------- epilogues ---

__global__ void __launch_bounds__(256) maxpool2x2_k(const float* __restrict__ in,
                                                    float* __restrict__ out,
                                                    int H, int W, int CB) {
  int H2 = H >> 1, W2 = W >> 1;
  int total = CB * H2 * W2;
  for (int idx = blockIdx.x * 256 + threadIdx.x; idx < total; idx += gridDim.x * 256) {
    int w2 = idx % W2; int t = idx / W2; int y2 = t % H2; int cb = t / H2;
    const float* p = in + ((size_t)cb * H + 2 * y2) * W + 2 * w2;
    out[idx] = fmaxf(fmaxf(p[0], p[1]), fmaxf(p[W], p[W + 1]));
  }
}

__global__ void __launch_bounds__(256) bn_stats_k(const float* __restrict__ x,
                                                  float* __restrict__ mv, int Nc) {
  int c = blockIdx.x;
  const float* p = x + (size_t)c * Nc;
  float s = 0.f, s2 = 0.f;
  for (int i = threadIdx.x; i < Nc; i += 256) { float v = p[i]; s += v; s2 += v * v; }
  __shared__ float sh[256], sh2[256];
  sh[threadIdx.x] = s; sh2[threadIdx.x] = s2; __syncthreads();
  for (int off = 128; off > 0; off >>= 1) {
    if (threadIdx.x < off) { sh[threadIdx.x] += sh[threadIdx.x + off];
                             sh2[threadIdx.x] += sh2[threadIdx.x + off]; }
    __syncthreads();
  }
  if (threadIdx.x == 0) {
    float m = sh[0] / Nc;
    mv[2 * c] = m;
    mv[2 * c + 1] = sh2[0] / Nc - m * m;   // biased var, like jnp.var
  }
}

// BN affine + activation quant; writes padded NHWC [B][H+2][W+2][C] (halo = 0).
// x (conv/pool output) is [C][B][H][W] with per-channel slice length Nc.
__global__ void __launch_bounds__(256) bn_apply_quant_k(const float* __restrict__ x,
                                                        const float* __restrict__ g,
                                                        const float* __restrict__ bt,
                                                        const float* __restrict__ mv,
                                                        const float* __restrict__ alpha,
                                                        _Float16* __restrict__ out,
                                                        int logC, int H, int W,
                                                        int Nc, int total) {
  __shared__ float s[NB + 1];
  if (alpha && threadIdx.x == 0) softmax7(alpha, s);
  __syncthreads();
  int C = 1 << logC, Wp = W + 2, Hp = H + 2;
  for (int idx = blockIdx.x * 256 + threadIdx.x; idx < total; idx += gridDim.x * 256) {
    int c = idx & (C - 1); int t = idx >> logC;
    int xp = t % Wp;  int t2 = t / Wp;
    int yp = t2 % Hp; int b = t2 / Hp;
    float o = 0.f;
    if (yp >= 1 && yp <= H && xp >= 1 && xp <= W) {
      float m = mv[2 * c], v = mv[2 * c + 1];
      float xv = x[(size_t)c * Nc + ((size_t)b * H + (yp - 1)) * W + (xp - 1)];
      float xn = (xv - m) * rsqrtf(v + 1e-5f) * g[c] + bt[c];
      if (alpha) o = mix_quant_act(xn, s);
      else {
        float xc = fminf(fmaxf(xn, 0.f), 1.f);
        o = rintf(xc * 255.f) * (1.f / 255.f);
      }
    }
    out[idx] = (_Float16)o;
  }
}

// ---------------------------------------------------------------- linear -----

// act: padded NHWC [B][6][6][256]; feature k = c*16 + y*4 + x
__global__ void __launch_bounds__(256) linear_out_k(const _Float16* __restrict__ act,
                                                    const _Float16* __restrict__ wq,
                                                    const float* __restrict__ bl,
                                                    float* __restrict__ out, int Bn) {
  int idx = blockIdx.x * 256 + threadIdx.x;
  if (idx >= Bn * 10) return;
  int b = idx / 10, j = idx - b * 10;
  float acc = bl[j];
  for (int k = 0; k < 4096; ++k) {
    int c = k >> 4, y = (k >> 2) & 3, x = k & 3;
    acc += (float)act[(((size_t)b * 6 + (y + 1)) * 6 + (x + 1)) * 256 + c] *
           (float)wq[j * 4096 + k];
  }
  out[idx] = acc;
}

// ------------------------------------------------------------------ host -----

extern "C" void kernel_launch(void* const* d_in, const int* in_sizes, int n_in,
                              void* d_out, int out_size, void* d_ws, size_t ws_size,
                              hipStream_t stream) {
  (void)in_sizes; (void)n_in; (void)out_size; (void)ws_size;
  const int Bn = 256;

  // inputs: 0:x, (1+3i):w_i, (2+3i):aa_i, (3+3i):aw_i, (19+2i):g_i, (20+2i):b_i,
  // 31:wl, 32:bl
  const float* x  = (const float*)d_in[0];
  const float* wl = (const float*)d_in[31];
  const float* bl = (const float*)d_in[32];

  char* ws = (char*)d_ws;
  size_t off = 0;
  auto carve = [&](size_t bytes) {
    void* p = ws + off;
    off = (off + bytes + 255) & ~(size_t)255;
    return p;
  };
  // max padded NHWC act: 256*34*34*64 = 18,939,904 f16
  _Float16* actA = (_Float16*)carve((size_t)18939904 * 2);
  _Float16* actB = (_Float16*)carve((size_t)18939904 * 2);
  float*    conv = (float*)   carve((size_t)16777216 * 4); // 64 * 262144 f32
  float*    poolb= (float*)   carve((size_t)4194304  * 4);
  _Float16* Aq   = (_Float16*)carve((size_t)2304 * 256 * 2);
  float*    mv   = (float*)   carve(512 * 4);
  float*    wmx  = (float*)   carve(256);

  const int CiT[6]   = {3, 64, 64, 128, 128, 256};
  const int CipT[6]  = {4, 64, 64, 128, 128, 256};
  const int shT[6]   = {2, 6, 6, 7, 7, 8};
  const int CoT[6]   = {64, 64, 128, 128, 256, 256};
  const int HT[6]    = {32, 32, 16, 16, 8, 8};
  const int logHT[6] = {5, 5, 4, 4, 3, 3};
  const int logCoT[6]= {6, 6, 7, 7, 8, 8};
  const bool PT[6]   = {false, true, false, true, false, true};

  // layer-0 input quantization (mix_act with aa0) -> padded NHWC, Cip=4
  {
    int total = Bn * 34 * 34 * 4;
    input_quant_k<<<(total + 255) / 256, 256, 0, stream>>>(
        x, (const float*)d_in[2], actA, Bn);
  }

  _Float16* cur = actA;
  for (int i = 0; i < 6; ++i) {
    const int Ci = CiT[i], Cip = CipT[i], ciShift = shT[i];
    const int Co = CoT[i], H = HT[i], W = HT[i], logW = logHT[i];
    const int K = 9 * Cip, Kpad = ((K + 31) / 32) * 32;
    const int Nn = Bn * H * W;
    const float* w  = (const float*)d_in[1 + 3 * i];
    const float* aw = (const float*)d_in[3 + 3 * i];
    const float* gg = (const float*)d_in[19 + 2 * i];
    const float* bb = (const float*)d_in[20 + 2 * i];
    const float* aa_next = (i < 5) ? (const float*)d_in[2 + 3 * (i + 1)] : nullptr;

    // mixed-bitwidth shared-weight quantization
    wtanh_max_k<<<1, 256, 0, stream>>>(w, Co * Ci * 9, wmx);
    {
      int total = Co * Kpad;
      wquant_mix_k<<<(total + 255) / 256, 256, 0, stream>>>(
          w, aw, wmx, Aq, Ci, Cip, ciShift, Kpad, Co);
    }

    // implicit-GEMM conv via WMMA
    dim3 grid(Nn / 64, Co / 16);
    if (i == 0)
      conv3x3_wmma_k<false><<<grid, 32, 0, stream>>>(cur, Aq, conv, logW, logW,
                                                     Cip, ciShift, K, Kpad, Nn);
    else
      conv3x3_wmma_k<true><<<grid, 32, 0, stream>>>(cur, Aq, conv, logW, logW,
                                                    Cip, ciShift, K, Kpad, Nn);

    const float* bnin = conv;
    int Ho = H, Wo = W;
    if (PT[i]) {
      Ho = H / 2; Wo = W / 2;
      int total = Co * Bn * Ho * Wo;
      maxpool2x2_k<<<(total + 255) / 256, 256, 0, stream>>>(conv, poolb, H, W, Co * Bn);
      bnin = poolb;
    }

    const int Nc = Bn * Ho * Wo;
    bn_stats_k<<<Co, 256, 0, stream>>>(bnin, mv, Nc);

    _Float16* nxt = (cur == actA) ? actB : actA;
    {
      int total = Bn * (Ho + 2) * (Wo + 2) * Co;
      int grid2 = (total + 255) / 256;
      bn_apply_quant_k<<<grid2, 256, 0, stream>>>(bnin, gg, bb, mv, aa_next, nxt,
                                                  logCoT[i], Ho, Wo, Nc, total);
    }
    cur = nxt;
  }

  // final 8-bit quantized linear: [256,4096] @ [10,4096]^T + bias
  wtanh_max_k<<<1, 256, 0, stream>>>(wl, 10 * 4096, wmx);
  wquant8_k<<<(10 * 4096 + 255) / 256, 256, 0, stream>>>(wl, wmx, Aq, 10 * 4096);
  linear_out_k<<<(Bn * 10 + 255) / 256, 256, 0, stream>>>(cur, Aq, bl, (float*)d_out, Bn);
}
